// SymmetricNN_6493990551583
// MI455X (gfx1250) — matrix operand order repaired and checked by reading, not compiled, not run on hardware
//
#include <hip/hip_runtime.h>
#include <hip/hip_fp16.h>

typedef __attribute__((ext_vector_type(16))) _Float16 v16h;
typedef __attribute__((ext_vector_type(8)))  _Float16 v8h;
typedef __attribute__((ext_vector_type(8)))  float    v8f;

#define TOTAL_F 0.078f
#define N_GRID 129
#define N_IN   72
#define N_Y    17
#define NCOLS  1224   // 72*17 (valid GEMM output columns)
#define NPAD   1232   // padded to 77*16
#define KPAD   128    // K=100 padded to 128
#define NTILES 77
#define OUTW   2193   // 129*17
#define NMIR   57
#define NMIRF  969    // 57*17

// fast sigmoid: v_exp_f32 + v_rcp_f32 (avoid IEEE div expansion)
__device__ __forceinline__ float sigmoidf(float v) {
    return __builtin_amdgcn_rcpf(1.0f + __expf(-v));
}

// ---- pack W3 [100,1224] f32 -> W3T [1232][128] f16 (K-contiguous, zero padded) ----
__global__ void pack_w3_kernel(const float* __restrict__ W3, _Float16* __restrict__ W3T) {
    int i = blockIdx.x * blockDim.x + threadIdx.x;
    if (i >= NPAD * KPAD) return;
    int n = i >> 7;      // column of original W3
    int k = i & 127;     // K index
    float v = 0.0f;
    if (n < NCOLS && k < 100) v = W3[k * NCOLS + n];
    W3T[i] = (_Float16)v;
}

// ---- mirror interpolation parameters (57 entries), faithful to the reference ----
__global__ void mirror_params_kernel(const float* __restrict__ xs,
                                     int* __restrict__ col1, int* __restrict__ col2,
                                     int* __restrict__ colnear, int* __restrict__ nearflag,
                                     float* __restrict__ wgt) {
    int j = threadIdx.x;
    if (j >= NMIR) return;
    float x_pos = TOTAL_F - xs[N_IN + j];
    int nr = (TOTAL_F - x_pos) < 0.02f ? 1 : 0;
    // argmin |x_pos - xs[i]| (first minimum), clipped to [0, N_IN-1]
    int best = 0; float bd = fabsf(x_pos - xs[0]);
    for (int i = 1; i < N_GRID; ++i) {
        float d = fabsf(x_pos - xs[i]);
        if (d < bd) { bd = d; best = i; }
    }
    if (best > N_IN - 1) best = N_IN - 1;
    // searchsorted(xs, x_pos, side='right') - 1, clipped to [0, N_IN-2]
    int p = 0;
    for (int i = 0; i < N_GRID; ++i) p += (xs[i] <= x_pos) ? 1 : 0;
    int c1 = p - 1;
    if (c1 < 0) c1 = 0;
    if (c1 > N_IN - 2) c1 = N_IN - 2;
    int c2 = c1 + 1;
    float w = (xs[c2] - x_pos) / (xs[c2] - xs[c1]);
    col1[j] = c1; col2[j] = c2; colnear[j] = best; nearflag[j] = nr; wgt[j] = w;
}

// ---- fused: layers 1+2 -> LDS, layer 3 via WMMA f16, sigmoid, store + mirror ----
__global__ __launch_bounds__(256) void fused_mlp_kernel(
    const float* __restrict__ x,  const float* __restrict__ W1, const float* __restrict__ b1,
    const float* __restrict__ W2, const float* __restrict__ b2, const float* __restrict__ b3,
    const _Float16* __restrict__ W3T,
    const int* __restrict__ col1, const int* __restrict__ col2,
    const int* __restrict__ colnear, const int* __restrict__ nearflag,
    const float* __restrict__ wgt,
    float* __restrict__ out)
{
    __shared__ float h1s[16 * 10];
    __shared__ __align__(16) _Float16 h2s[16 * KPAD];   // A operand staging (f16)
    __shared__ __align__(16) _Float16 xgs[16 * NPAD];   // post-sigmoid tile (f16) for mirror

    const int tid = threadIdx.x;
    const int rowBase = blockIdx.x * 16;
    float* outBase = out + (size_t)rowBase * OUTW;

    // ---- layer 1: h1 = sigmoid(x*W1 + b1), 16 rows x 10 ----
    if (tid < 160) {
        int r = tid / 10, k = tid - r * 10;
        float xv = x[rowBase + r];
        h1s[tid] = sigmoidf(xv * W1[k] + b1[k]);
    }
    __syncthreads();

    // ---- layer 2: h2 = sigmoid(h1@W2 + b2) -> f16 [16][128] (zero padded K) ----
    #pragma unroll
    for (int i = 0; i < 8; ++i) {
        int slot = i * 256 + tid;         // 16*128 = 2048 slots
        int r = slot >> 7, j = slot & 127;
        float v = 0.0f;
        if (j < 100) {
            float acc = b2[j];
            #pragma unroll
            for (int k = 0; k < 10; ++k) acc += h1s[r * 10 + k] * W2[k * 100 + j];
            v = sigmoidf(acc);
        }
        h2s[slot] = (_Float16)v;
    }
    __syncthreads();

    const int lane   = tid & 31;
    const int waveId = tid >> 5;
    const int hl = lane & 15;     // column within tile / A row
    const int hi = lane >> 4;     // half-wave select

    // ---- A fragments (16x32 f16 layout): shared across all N tiles of this wave ----
    // lanes 0-15: row=hl, K = kc*32 + {0..7, 16..23}; lanes 16-31: K = kc*32 + {8..15, 24..31}
    v16h afrag[4];
    #pragma unroll
    for (int kc = 0; kc < 4; ++kc) {
        int k0 = kc * 32 + hi * 8;
        v8h lo  = *(const v8h*)&h2s[hl * KPAD + k0];
        v8h hi8 = *(const v8h*)&h2s[hl * KPAD + k0 + 16];
        afrag[kc] = __builtin_shufflevector(lo, hi8, 0,1,2,3,4,5,6,7,8,9,10,11,12,13,14,15);
    }

    // ---- layer 3 GEMM: each wave walks N tiles t = waveId, waveId+8, ... (wave-uniform) ----
    for (int t = waveId; t < NTILES; t += 8) {
        int n = t * 16 + hl;      // output column for this lane
        v8f acc = {};
        #pragma unroll
        for (int kc = 0; kc < 4; ++kc) {
            // B fragment (32x16 f16): lane holds col=hl, K = kc*32 + hi*8 + {0..7, 16..23}
            const v8h blo = *(const v8h*)&W3T[n * KPAD + kc * 32 + hi * 8];
            const v8h bhi = *(const v8h*)&W3T[n * KPAD + kc * 32 + hi * 8 + 16];
            const v16h bfrag = __builtin_shufflevector(blo, bhi, 0,1,2,3,4,5,6,7,8,9,10,11,12,13,14,15);
            acc = __builtin_amdgcn_wmma_f32_16x16x32_f16(
                false, afrag[kc], false, bfrag, (short)0, acc, false, false);
        }
        float bias = (n < NCOLS) ? b3[n] : 0.0f;
        #pragma unroll
        for (int v = 0; v < 8; ++v) {
            int rl = v + hi * 8;                  // local row (C/D layout)
            float s = sigmoidf(acc[v] + bias);
            xgs[rl * NPAD + n] = (_Float16)s;     // keep for mirror phase
            if (n < NCOLS) outBase[rl * OUTW + n] = s;   // xg part of output (f32)
        }
    }
    __syncthreads();

    // ---- mirrored columns out[:, 1224:2193] from LDS (no HBM re-read) ----
    for (int i = tid; i < 16 * NMIRF; i += 256) {
        int r = i / NMIRF;
        int q = i - r * NMIRF;
        int m = q / N_Y;
        int y = q - m * N_Y;
        float val;
        if (nearflag[m]) {
            val = (float)xgs[r * NPAD + colnear[m] * N_Y + y];
        } else {
            float w = wgt[m];
            val = w * (float)xgs[r * NPAD + col1[m] * N_Y + y]
                + (1.0f - w) * (float)xgs[r * NPAD + col2[m] * N_Y + y];
        }
        outBase[r * OUTW + NCOLS + q] = val;
    }
}

extern "C" void kernel_launch(void* const* d_in, const int* in_sizes, int n_in,
                              void* d_out, int out_size, void* d_ws, size_t ws_size,
                              hipStream_t stream) {
    const float* x  = (const float*)d_in[0];
    const float* W1 = (const float*)d_in[1];
    const float* b1 = (const float*)d_in[2];
    const float* W2 = (const float*)d_in[3];
    const float* b2 = (const float*)d_in[4];
    const float* W3 = (const float*)d_in[5];
    const float* b3 = (const float*)d_in[6];
    const float* xs = (const float*)d_in[7];
    float* out = (float*)d_out;

    char* ws = (char*)d_ws;
    int*      c1  = (int*)(ws + 0);
    int*      c2  = (int*)(ws + 256);
    int*      cn  = (int*)(ws + 512);
    int*      nf  = (int*)(ws + 768);
    float*    wg  = (float*)(ws + 1024);
    _Float16* W3T = (_Float16*)(ws + 1536);   // 1232*128*2 = 315392 bytes

    const int B = in_sizes[0];   // 65536 rows

    pack_w3_kernel<<<(NPAD * KPAD + 255) / 256, 256, 0, stream>>>(W3, W3T);
    mirror_params_kernel<<<1, 64, 0, stream>>>(xs, c1, c2, cn, nf, wg);
    fused_mlp_kernel<<<B / 16, 256, 0, stream>>>(x, W1, b1, W2, b2, b3, W3T,
                                                 c1, c2, cn, nf, wg, out);
}
